// SSIM_62208306315856
// MI455X (gfx1250) — compile-verified, matrix operand order checked by reading, not run on hardware
//
#include <hip/hip_runtime.h>
#include <math.h>

typedef __attribute__((ext_vector_type(16))) _Float16 v16h;
typedef __attribute__((ext_vector_type(8)))  float    v8f;

#define WMMA_F16(a, b, c) \
  __builtin_amdgcn_wmma_f32_16x16x32_f16(false, (a), false, (b), (short)0, (c), false, false)

// --- helpers -----------------------------------------------------------------

__device__ __forceinline__ v16h mulh(v16h a, v16h b) {
  v16h r;
#pragma unroll
  for (int e = 0; e < 16; ++e) r[e] = a[e] * b[e];
  return r;
}

// Masked load for the right-edge tile column (c0+21 may exceed 511).
// ISA 16-bit A layout: lane = row M=l&15; element e -> K = 8*hi+e (e<8)
// or 16+8*hi+(e-8). Zero outside the 22 needed columns / image bounds.
__device__ __forceinline__ v16h load_frag_edge(const float* __restrict__ img,
                                               int row, int c0, int hi) {
  v16h a;
  bool rok = (row < 512);
  const float* rp = img + (size_t)row * 512;
#pragma unroll
  for (int e = 0; e < 16; ++e) {
    int K = (e < 8) ? (8 * hi + e) : (8 + 8 * hi + e);
    int col = c0 + K;
    float v = (rok && K < 22 && col < 512) ? rp[col] : 0.0f;
    a[e] = (_Float16)v;
  }
  return a;
}

// Fast interior load: c0 is 64B-aligned and c0+21 < 512, so all column
// accesses are unconditional aligned vector loads. Only the row needs a check.
__device__ __forceinline__ v16h load_frag_fast(const float* __restrict__ img,
                                               int row, int c0, int hi) {
  v16h a;
  if (row < 512) {
    const float* rp = img + (size_t)row * 512 + c0;
    float4 u0 = *(const float4*)(rp + 8 * hi);      // cols K = 8*hi + 0..3
    float4 u1 = *(const float4*)(rp + 8 * hi + 4);  // cols K = 8*hi + 4..7
    a[0] = (_Float16)u0.x; a[1] = (_Float16)u0.y;
    a[2] = (_Float16)u0.z; a[3] = (_Float16)u0.w;
    a[4] = (_Float16)u1.x; a[5] = (_Float16)u1.y;
    a[6] = (_Float16)u1.z; a[7] = (_Float16)u1.w;
    if (hi == 0) {                                  // cols K = 16..21
      float4 u2 = *(const float4*)(rp + 16);
      float2 u3 = *(const float2*)(rp + 20);
      a[8]  = (_Float16)u2.x; a[9]  = (_Float16)u2.y;
      a[10] = (_Float16)u2.z; a[11] = (_Float16)u2.w;
      a[12] = (_Float16)u3.x; a[13] = (_Float16)u3.y;
    } else {
#pragma unroll
      for (int e = 8; e < 14; ++e) a[e] = (_Float16)0.0f;
    }
    a[14] = (_Float16)0.0f; a[15] = (_Float16)0.0f;
  } else {
#pragma unroll
    for (int e = 0; e < 16; ++e) a[e] = (_Float16)0.0f;
  }
  return a;
}

__device__ __forceinline__ v16h load_frag(const float* __restrict__ img,
                                          int row, int c0, int hi, bool edge) {
  return edge ? load_frag_edge(img, row, c0, hi) : load_frag_fast(img, row, c0, hi);
}

// Separable 7-tap conv on a 16x16 output tile:
//   T = [A1;A2](22x32 padded) * Bg(32x16 Toeplitz)   (2 horizontal WMMAs)
//   out = G(16x32 Toeplitz) * T(32x16)               (1 vertical WMMA)
// f32 C-fragments convert lane-locally into the vertical B-fragment:
//   C layout lane holds M = r + 8*hi in VGPR r; B layout needs K = 8*hi + e
//   (from t1[e]) and K = 16 + 8*hi + (e-8) (from t2[e-8]). No masking needed:
//   gfrag columns K>=22 are all zero and t1/t2 are always finite, so the
//   T-rows >= 22 that land in vb contribute exactly zero.
__device__ __forceinline__ v8f conv_sep(v16h a1, v16h a2, v16h bfrag, v16h gfrag) {
  v8f z = {};
  v8f t1 = WMMA_F16(a1, bfrag, z);
  v8f t2 = WMMA_F16(a2, bfrag, z);
  v16h vb;
#pragma unroll
  for (int e = 0; e < 8; ++e) vb[e] = (_Float16)t1[e];
#pragma unroll
  for (int e = 0; e < 8; ++e) vb[8 + e] = (_Float16)t2[e];
  return WMMA_F16(gfrag, vb, z);
}

__device__ __forceinline__ float ssim_term(float m1, float m2, float xx, float yy,
                                           float xy) {
  const float C1 = 1e-4f, C2 = 9e-4f;
  float m1s = m1 * m1, m2s = m2 * m2, m12 = m1 * m2;
  float s1  = xx - m1s;
  float s2  = yy - m2s;
  float s12 = xy - m12;
  float num = (2.0f * m12 + C1) * (2.0f * s12 + C2);
  float den = (m1s + m2s + C1) * (s1 + s2 + C2);
  return 1.0f - num * __builtin_amdgcn_rcpf(den);   // den >= C1*C2 > 0
}

// --- main kernel: one wave per strip of 8 vertically-adjacent 16x16 tiles ----

__global__ __launch_bounds__(32) void ssim_strip_kernel(
    const float* __restrict__ gt, const float* __restrict__ ny,
    const float* __restrict__ w, float* __restrict__ partial) {
  const int l     = threadIdx.x;
  const int l15   = l & 15;
  const int hi    = l >> 4;
  const int c0    = blockIdx.x * 16;       // output col base (strip column)
  const int strip = blockIdx.y;            // 0..3, 8 tiles each
  const int img   = blockIdx.z;            // (b*C + c), contiguous layout
  const bool edge = (c0 + 21 >= 512);      // only tile column 31
  const bool col_ok = (c0 + 15 < 506);     // whole tile column inside output

  // 1-D Gaussian factors from the supplied (separable) 2-D window.
  float s = sqrtf(w[24]);
  float gh[7], gvv[7];
#pragma unroll
  for (int j = 0; j < 7; ++j) {
    gh[j]  = w[21 + j] / s;
    gvv[j] = w[7 * j + 3] / s;
  }

  // Tile-invariant Toeplitz fragments (built once per wave).
  // B: Bg[K][n] = gh[K-n]; A(vertical): G[m][K] = gvv[K-m]; n/m = l15.
  v16h bfrag, gfrag;
#pragma unroll
  for (int e = 0; e < 16; ++e) {
    int K = (e < 8) ? (8 * hi + e) : (8 + 8 * hi + e);
    int d = K - l15;
    bool in = (d >= 0) && (d <= 6);
    bfrag[e] = in ? (_Float16)gh[d]  : (_Float16)0.0f;
    gfrag[e] = in ? (_Float16)gvv[d] : (_Float16)0.0f;
  }

  const float* px = gt + (size_t)img * (512 * 512);
  const float* py = ny + (size_t)img * (512 * 512);

  int r0 = strip * 128;                    // first tile's output row base

  // Fragments for rows [r0, r0+16) and [r0+16, r0+32); the second becomes the
  // first of the next tile (identical lane layout), so each row is loaded once.
  v16h a1x = load_frag(px, r0 + l15,      c0, hi, edge);
  v16h a1y = load_frag(py, r0 + l15,      c0, hi, edge);
  v16h a2x = load_frag(px, r0 + 16 + l15, c0, hi, edge);
  v16h a2y = load_frag(py, r0 + 16 + l15, c0, hi, edge);
  v16h p1xx = mulh(a1x, a1x), p1yy = mulh(a1y, a1y), p1xy = mulh(a1x, a1y);
  v16h p2xx = mulh(a2x, a2x), p2yy = mulh(a2y, a2y), p2xy = mulh(a2x, a2y);

  float sum = 0.0f;

  for (int t = 0; t < 8; ++t) {
    // Prefetch the next tile's new fragments before this tile's matrix work.
    v16h nx, nyf;
    if (t < 7) {
      nx  = load_frag(px, r0 + 32 + l15, c0, hi, edge);
      nyf = load_frag(py, r0 + 32 + l15, c0, hi, edge);
    }

    v8f mu1 = conv_sep(a1x,  a2x,  bfrag, gfrag);
    v8f mu2 = conv_sep(a1y,  a2y,  bfrag, gfrag);
    v8f exx = conv_sep(p1xx, p2xx, bfrag, gfrag);
    v8f eyy = conv_sep(p1yy, p2yy, bfrag, gfrag);
    v8f exy = conv_sep(p1xy, p2xy, bfrag, gfrag);

    if (col_ok && (r0 + 15 < 506)) {
      // Interior tile: every element of the 16x16 output is valid.
#pragma unroll
      for (int r = 0; r < 8; ++r)
        sum += ssim_term(mu1[r], mu2[r], exx[r], eyy[r], exy[r]);
    } else {
      // Boundary tile: mask to the valid 506x506 region.
#pragma unroll
      for (int r = 0; r < 8; ++r) {
        int om = r0 + r + 8 * hi;          // C/D layout: VGPR r -> M = r + 8*hi
        int on = c0 + l15;
        if (om < 506 && on < 506)
          sum += ssim_term(mu1[r], mu2[r], exx[r], eyy[r], exy[r]);
      }
    }

    if (t < 7) {                           // slide the strip window down 16 rows
      a1x = a2x;  a1y = a2y;
      p1xx = p2xx; p1yy = p2yy; p1xy = p2xy;
      a2x = nx;   a2y = nyf;
      p2xx = mulh(a2x, a2x); p2yy = mulh(a2y, a2y); p2xy = mulh(a2x, a2y);
      r0 += 16;
    }
  }

  // Wave32 butterfly reduction; one deterministic partial per strip.
#pragma unroll
  for (int off = 16; off > 0; off >>= 1) sum += __shfl_xor(sum, off, 32);
  if (l == 0)
    partial[((size_t)img * gridDim.y + strip) * gridDim.x + blockIdx.x] = sum;
}

// --- final reduction: single block, fixed order -> deterministic -------------

__global__ __launch_bounds__(256) void ssim_reduce_kernel(
    const float* __restrict__ partial, float* __restrict__ out, int n, float scale) {
  __shared__ float sm[256];
  float s = 0.0f;
  for (int i = threadIdx.x; i < n; i += 256) s += partial[i];
  sm[threadIdx.x] = s;
  __syncthreads();
#pragma unroll
  for (int k = 128; k > 0; k >>= 1) {
    if ((int)threadIdx.x < k) sm[threadIdx.x] += sm[threadIdx.x + k];
    __syncthreads();
  }
  if (threadIdx.x == 0) out[0] = sm[0] * scale;
}

// --- launch ------------------------------------------------------------------

extern "C" void kernel_launch(void* const* d_in, const int* in_sizes, int n_in,
                              void* d_out, int out_size, void* d_ws, size_t ws_size,
                              hipStream_t stream) {
  const float* gt = (const float*)d_in[0];   // (32,3,512,512) f32
  const float* ny = (const float*)d_in[1];   // (32,3,512,512) f32
  const float* w  = (const float*)d_in[2];   // (3,1,7,7) f32, identical per channel
  float* out      = (float*)d_out;           // scalar f32
  float* partial  = (float*)d_ws;            // 32*4*96 f32 partial sums

  dim3 grid(32, 4, 96);                      // strip cols, strips of 8 tiles, images
  ssim_strip_kernel<<<grid, 32, 0, stream>>>(gt, ny, w, partial);

  const int n = 32 * 4 * 96;                 // 12288 partials
  const float scale = 1.0f / (32.0f * 3.0f * 506.0f * 506.0f);
  ssim_reduce_kernel<<<1, 256, 0, stream>>>(partial, out, n, scale);
}
